// GRULayer_16527034155416
// MI455X (gfx1250) — matrix operand order repaired and checked by reading, not compiled, not run on hardware
//
#include <hip/hip_runtime.h>
#include <hip/hip_bf16.h>

// ---------------------------------------------------------------------------
// GRU layer, T=2048, B=32, V=512, H=512 on gfx1250 (MI455X).
// Architecture: weights pinned in LDS (192KB/WGP across 16 persistent WGs,
// filled via TDM tensor_load_to_lds), bf16 WMMA (v_wmma_f32_16x16x32_bf16)
// with f32 accumulation and multi-chain accumulators for WMMA latency hiding,
// recurrence exchanged through L2-resident workspace, 2 grid barriers/step.
// ---------------------------------------------------------------------------

#define T_STEPS 2048
#define BATCH   32
#define VDIM    512
#define HDIM    512
#define NWG     16          // one WG per 32-column slice of H
#define NTHREADS 128        // 4 waves; wave = one 16x16 output tile
#define WG_W_ELEMS 98304    // 3 gates * 2 ntiles * 32 kchunks * 512 bf16
#define WG_W_BYTES (WG_W_ELEMS * 2)   // 192 KB LDS per WG

typedef __attribute__((ext_vector_type(16))) __bf16 v16bf;
typedef __attribute__((ext_vector_type(8)))  float  v8f;
typedef __attribute__((ext_vector_type(4)))  unsigned int v4u;
typedef __attribute__((ext_vector_type(8)))  int  v8i;
typedef __attribute__((ext_vector_type(4)))  int  v4i;

#if defined(__has_builtin)
#  if __has_builtin(__builtin_amdgcn_tensor_load_to_lds) && \
      __has_builtin(__builtin_amdgcn_s_wait_tensorcnt)
#    define USE_TDM 1
#  endif
#endif
#ifndef USE_TDM
#  define USE_TDM 0
#endif

union BfVec { uint4 q[2]; v16bf v; };

__device__ __forceinline__ unsigned short f2bf(float f) {
  unsigned u = __float_as_uint(f);
  unsigned r = u + 0x7FFFu + ((u >> 16) & 1u);   // round-to-nearest-even
  return (unsigned short)(r >> 16);
}

__device__ __forceinline__ float sigmoidf_(float x) {
  return 1.0f / (1.0f + __expf(-x));
}

__device__ __forceinline__ float tanhf_(float x) {
  float e = __expf(-2.0f * fabsf(x));
  float t = (1.0f - e) / (1.0f + e);
  return copysignf(t, x);
}

// Sense-free monotonic grid barrier over 16 co-resident WGs.
__device__ __forceinline__ void grid_barrier(unsigned* bar) {
  __syncthreads();
  if (threadIdx.x == 0) {
    __threadfence();                              // release
    unsigned arr    = atomicAdd(bar, 1u);
    unsigned target = (arr / NWG + 1u) * NWG;
    while (atomicAdd(bar, 0u) < target) {
      __builtin_amdgcn_s_sleep(1);
    }
    __threadfence();                              // acquire
  }
  __syncthreads();
}

// One K-segment (16 chunks of K=32) of a 16x16 tile MAC, accumulated into
// TWO independent WMMA chains (even/odd kc) to hide WMMA RAW latency.
// A: row-major bf16 [32][512] in global (L2-hot). W: LDS, B-ready layout.
__device__ __forceinline__ void seg_mma(v8f& acc0, v8f& acc1,
                                        const unsigned short* A,
                                        int mt, const uint4* W, int lane) {
  const unsigned short* arow =
      A + (mt * 16 + (lane & 15)) * HDIM + ((lane & 16) ? 8 : 0);
  const uint4* wl = W + lane * 2;
  BfVec a0, a1, b0, b1;
#pragma unroll
  for (int kc = 0; kc < 16; kc += 2) {
    a0.q[0] = *(const uint4*)(arow + kc * 32);          // K = base..base+7
    a0.q[1] = *(const uint4*)(arow + kc * 32 + 16);     // K = base+16..+23
    b0.q[0] = wl[kc * 64];
    b0.q[1] = wl[kc * 64 + 1];
    a1.q[0] = *(const uint4*)(arow + kc * 32 + 32);
    a1.q[1] = *(const uint4*)(arow + kc * 32 + 48);
    b1.q[0] = wl[kc * 64 + 64];
    b1.q[1] = wl[kc * 64 + 65];
    acc0 = __builtin_amdgcn_wmma_f32_16x16x32_bf16(
        false, a0.v, false, b0.v, (short)0, acc0, false, false);
    acc1 = __builtin_amdgcn_wmma_f32_16x16x32_bf16(
        false, a1.v, false, b1.v, (short)0, acc1, false, false);
  }
}

// Same A, two weight sets (candidate + update gates): one A load feeds the
// two gate chains -> 2 independent WMMA chains per call (4 across both accs).
__device__ __forceinline__ void seg_mma2(v8f& accN0, v8f& accN1,
                                         v8f& accZ0, v8f& accZ1,
                                         const unsigned short* A, int mt,
                                         const uint4* WN, const uint4* WZ,
                                         int lane) {
  const unsigned short* arow =
      A + (mt * 16 + (lane & 15)) * HDIM + ((lane & 16) ? 8 : 0);
  const uint4* wn = WN + lane * 2;
  const uint4* wz = WZ + lane * 2;
  BfVec a, bn, bz;
#pragma unroll
  for (int kc = 0; kc < 16; ++kc) {
    a.q[0]  = *(const uint4*)(arow + kc * 32);
    a.q[1]  = *(const uint4*)(arow + kc * 32 + 16);
    bn.q[0] = wn[kc * 64];
    bn.q[1] = wn[kc * 64 + 1];
    bz.q[0] = wz[kc * 64];
    bz.q[1] = wz[kc * 64 + 1];
    if (kc & 1) {
      accN1 = __builtin_amdgcn_wmma_f32_16x16x32_bf16(
          false, a.v, false, bn.v, (short)0, accN1, false, false);
      accZ1 = __builtin_amdgcn_wmma_f32_16x16x32_bf16(
          false, a.v, false, bz.v, (short)0, accZ1, false, false);
    } else {
      accN0 = __builtin_amdgcn_wmma_f32_16x16x32_bf16(
          false, a.v, false, bn.v, (short)0, accN0, false, false);
      accZ0 = __builtin_amdgcn_wmma_f32_16x16x32_bf16(
          false, a.v, false, bz.v, (short)0, accZ0, false, false);
    }
  }
}

__global__ void init_bar_kernel(unsigned* bar) {
  if (threadIdx.x < 8) bar[threadIdx.x] = 0u;
}

// Pre-swizzle fp32 weights into bf16 "B-ready" layout:
// wsW[wg][g][nt][kc][lane][j] = W_g[k][n], n = wg*32+nt*16+(lane&15),
// k = kc*32 + ((lane&16)?8:0) + (j<8 ? j : j+8).  Table K order per lane.
__global__ void prep_w_kernel(const float* __restrict__ Wr,
                              const float* __restrict__ Wn,
                              const float* __restrict__ Wz,
                              unsigned short* __restrict__ wsW) {
  int idx = blockIdx.x * 256 + threadIdx.x;       // total NWG*98304
  int wg  = idx / WG_W_ELEMS;
  int rem = idx % WG_W_ELEMS;
  int g   = rem / 32768;  rem %= 32768;
  int nt  = rem / 16384;  rem %= 16384;
  int kc  = rem / 512;    rem %= 512;
  int lane = rem / 16;
  int j    = rem % 16;
  int n = wg * 32 + nt * 16 + (lane & 15);
  int k = kc * 32 + ((lane & 16) ? 8 : 0) + ((j < 8) ? j : j + 8);
  const float* W = (g == 0) ? Wr : (g == 1) ? Wn : Wz;
  wsW[idx] = f2bf(W[k * HDIM + n]);
}

__global__ void __launch_bounds__(NTHREADS)
gru_persist_kernel(const float* __restrict__ inputs,   // [T][32][512] f32
                   const float* __restrict__ state,    // [1][32][512] f32
                   const float* __restrict__ b_reset,
                   const float* __restrict__ b_net,
                   const float* __restrict__ b_update,
                   const unsigned short* __restrict__ wsW,
                   unsigned short* __restrict__ xA,    // [2][32][512] bf16
                   unsigned short* __restrict__ hA,    // [32][512] bf16
                   unsigned short* __restrict__ hrA,   // [32][512] bf16
                   unsigned* __restrict__ bar,
                   float* __restrict__ out) {          // [T][32][512] + [32][512]
  extern __shared__ uint4 smemv[];                     // 192 KB weights

  const int wg   = blockIdx.x;
  const int tid  = threadIdx.x;
  const int wave = tid >> 5;
  const int lane = tid & 31;

  // ---- Prologue: pin this WG's weight slice into LDS (once, reused 2048x).
#if USE_TDM
  if (wave == 0) {
    // Tensor DMA descriptor (D#): 1-D tile of 24576 x 8-byte elements
    // (192 KB) from wsW + wg*192KB into LDS offset 0.
    const unsigned long long ga =
        (unsigned long long)(wsW + (size_t)wg * WG_W_ELEMS);
    const unsigned nelem = WG_W_BYTES / 8;           // 24576 8B elements
    v4u g0;
    g0.x = 1u;                                       // count=1 (valid user D#)
    g0.y = 0u;                                       // lds_addr = 0
    g0.z = (unsigned)(ga & 0xFFFFFFFFu);             // global_addr[31:0]
    g0.w = (unsigned)((ga >> 32) & 0x01FFFFFFu)      // global_addr[56:32]
         | 0x80000000u;                              // type = 2 ("image")
    v8i g1;
    g1[0] = 0x00030000;                              // data_size=3 (8B), mask=0
    g1[1] = (int)(nelem << 16);                      // tensor_dim0[15:0]
    g1[2] = (int)((nelem >> 16) | (1u << 16));       // tensor_dim0[31:16], tensor_dim1=1
    g1[3] = (int)(nelem << 16);                      // tile_dim0 = 24576
    g1[4] = 0;                                       // tile_dim1/2 unused
    g1[5] = (int)nelem;                              // tensor_dim0_stride[31:0]
    g1[6] = 0;
    g1[7] = 0;
    v4i g2; g2[0] = 1; g2[1] = 1; g2[2] = 0; g2[3] = 0;  // tensor_dim2/3 = 1
    v4i g3; g3[0] = 0; g3[1] = 0x00010000; g3[2] = 0; g3[3] = 0; // tensor_dim4=1
    v8i g4; g4[0] = 0; g4[1] = 0; g4[2] = 0; g4[3] = 0;  // unused trailing group
    g4[4] = 0; g4[5] = 0; g4[6] = 0; g4[7] = 0;          // (6-arg builtin form)
    __builtin_amdgcn_tensor_load_to_lds(g0, g1, g2, g3, g4, 0);
    __builtin_amdgcn_s_wait_tensorcnt(0);
  }
#else
  {
    const uint4* src = (const uint4*)wsW + (size_t)wg * (WG_W_ELEMS / 8);
    for (int i = tid; i < WG_W_ELEMS / 8; i += NTHREADS) smemv[i] = src[i];
  }
#endif

  // Tile assignment: wave -> (mt, nt); fixed for whole kernel.
  const int mt = wave & 1;
  const int nt = wave >> 1;
  const int colLocal = lane & 15;
  const int col   = wg * 32 + nt * 16 + colLocal;      // H column this lane owns
  const int rbase = mt * 16 + ((lane & 16) ? 8 : 0);   // first of 8 batch rows
  const float brv = b_reset[col];
  const float bnv = b_net[col];
  const float bzv = b_update[col];

  // LDS weight bases (uint4 units) per (gate, seg).
  const uint4* Wr_x = smemv + ((0 * 2 + nt) * 32 + 0)  * 64;
  const uint4* Wr_h = smemv + ((0 * 2 + nt) * 32 + 16) * 64;
  const uint4* Wn_x = smemv + ((1 * 2 + nt) * 32 + 0)  * 64;
  const uint4* Wn_h = smemv + ((1 * 2 + nt) * 32 + 16) * 64;
  const uint4* Wz_x = smemv + ((2 * 2 + nt) * 32 + 0)  * 64;
  const uint4* Wz_h = smemv + ((2 * 2 + nt) * 32 + 16) * 64;

  // Register-carried h for the 8 (row,col) cells this lane owns; init from state.
  float hprev[8];
#pragma unroll
  for (int j = 0; j < 8; ++j) hprev[j] = state[(rbase + j) * HDIM + col];

  // Initialize shared hA (bf16) for this WG's column slice.
  {
    int r  = tid >> 2;            // 32 rows
    int c8 = (tid & 3) * 8;       // 4 x 8 cols = 32-col slice
    const float* hs = state + r * HDIM + wg * 32 + c8;
    uint4 p;
    p.x = f2bf(hs[0]) | ((unsigned)f2bf(hs[1]) << 16);
    p.y = f2bf(hs[2]) | ((unsigned)f2bf(hs[3]) << 16);
    p.z = f2bf(hs[4]) | ((unsigned)f2bf(hs[5]) << 16);
    p.w = f2bf(hs[6]) | ((unsigned)f2bf(hs[7]) << 16);
    *(uint4*)(hA + r * HDIM + wg * 32 + c8) = p;
  }
  __syncthreads();   // weights in LDS + hA staged

  float* outF = out + (size_t)T_STEPS * BATCH * HDIM;

  for (int t = 0; t < T_STEPS; ++t) {
    const int par = t & 1;
    unsigned short* xbuf = xA + par * (BATCH * VDIM);

    // ---- Stage 1: cooperative fp32 -> bf16 convert of x(t), double-buffered.
    {
      const size_t e = (size_t)wg * 1024 + (size_t)tid * 8;
      const float* xs = inputs + (size_t)t * BATCH * VDIM + e;
      float4 f0 = ((const float4*)xs)[0];
      float4 f1 = ((const float4*)xs)[1];
      uint4 p;
      p.x = f2bf(f0.x) | ((unsigned)f2bf(f0.y) << 16);
      p.y = f2bf(f0.z) | ((unsigned)f2bf(f0.w) << 16);
      p.z = f2bf(f1.x) | ((unsigned)f2bf(f1.y) << 16);
      p.w = f2bf(f1.z) | ((unsigned)f2bf(f1.w) << 16);
      *(uint4*)(xbuf + e) = p;
      if (t + 1 < T_STEPS)    // warm L2 for next step's x tile
        __builtin_prefetch(inputs + (size_t)(t + 1) * BATCH * VDIM + e, 0, 1);
    }

    grid_barrier(bar);   // x(t) + h(t-1) visible everywhere

    // ---- Stage 2: reset gate  R = sigmoid([x,h] @ W_r + b_r), hr = h*R
    {
      v8f a0 = {}, a1 = {};
      seg_mma(a0, a1, xbuf, mt, Wr_x, lane);
      seg_mma(a0, a1, hA,   mt, Wr_h, lane);
      v8f acc = a0 + a1;
#pragma unroll
      for (int j = 0; j < 8; ++j) {
        float R   = sigmoidf_(acc[j] + brv);
        float hrv = hprev[j] * R;
        hrA[(rbase + j) * HDIM + col] = f2bf(hrv);
      }
    }

    grid_barrier(bar);   // hr(t) visible everywhere

    // ---- Stage 3: candidate + update gates, fused (A loaded once, 4 chains).
    {
      v8f aN0 = {}, aN1 = {}, aZ0 = {}, aZ1 = {};
      seg_mma2(aN0, aN1, aZ0, aZ1, xbuf, mt, Wn_x, Wz_x, lane);
      seg_mma2(aN0, aN1, aZ0, aZ1, hrA,  mt, Wn_h, Wz_h, lane);
      v8f aN = aN0 + aN1;
      v8f aZ = aZ0 + aZ1;
      float* orow = out + (size_t)t * BATCH * HDIM;
#pragma unroll
      for (int j = 0; j < 8; ++j) {
        float Hc = tanhf_(aN[j] + bnv);
        float Z  = sigmoidf_(aZ[j] + bzv);
        float hn = Z * hprev[j] + (1.0f - Z) * Hc;
        hprev[j] = hn;
        int off = (rbase + j) * HDIM + col;
        orow[off] = hn;                 // states[t]
        hA[off]   = f2bf(hn);           // h for next step's reset GEMM
        if (t == T_STEPS - 1) outF[off] = hn;   // h_final
      }
    }
  }
}

extern "C" void kernel_launch(void* const* d_in, const int* in_sizes, int n_in,
                              void* d_out, int out_size, void* d_ws, size_t ws_size,
                              hipStream_t stream) {
  (void)in_sizes; (void)n_in; (void)out_size; (void)ws_size;
  const float* inputs = (const float*)d_in[0];
  const float* state  = (const float*)d_in[1];
  const float* Wr     = (const float*)d_in[2];
  const float* br     = (const float*)d_in[3];
  const float* Wn     = (const float*)d_in[4];
  const float* bn     = (const float*)d_in[5];
  const float* Wz     = (const float*)d_in[6];
  const float* bz     = (const float*)d_in[7];
  float* out = (float*)d_out;

  // Workspace layout (all 256B-aligned): swizzled weights, x double buffer,
  // h, h*R exchange buffers, grid-barrier word.  Total ~3.3 MB.
  char* ws = (char*)d_ws;
  unsigned short* wsW = (unsigned short*)ws;                       // 3,145,728 B
  unsigned short* xA  = (unsigned short*)(ws + 3145728);           //    65,536 B
  unsigned short* hA  = (unsigned short*)(ws + 3145728 + 65536);   //    32,768 B
  unsigned short* hrA = (unsigned short*)(ws + 3145728 + 98304);   //    32,768 B
  unsigned*       bar = (unsigned*)(ws + 3145728 + 131072);

  (void)hipFuncSetAttribute((const void*)gru_persist_kernel,
                            hipFuncAttributeMaxDynamicSharedMemorySize,
                            WG_W_BYTES);

  init_bar_kernel<<<1, 32, 0, stream>>>(bar);
  prep_w_kernel<<<(NWG * WG_W_ELEMS) / 256, 256, 0, stream>>>(Wr, Wn, Wz, wsW);
  gru_persist_kernel<<<NWG, NTHREADS, WG_W_BYTES, stream>>>(
      inputs, state, br, bn, bz, wsW, xA, hA, hrA, bar, out);
}
